// MoELayer_50216757625332
// MI455X (gfx1250) — matrix operand order repaired and checked by reading, not compiled, not run on hardware
//
#include <hip/hip_runtime.h>

// ---------------------------------------------------------------------------
// MoE dense dispatch for MI455X (gfx1250, wave32, WMMA).
//   out[b,d] = sum_e probs[e] * ( GELU(h @ W1[e] + b1[e]) @ W2[e] + b2[e] )
// f16 WMMA (v_wmma_f32_16x16x32_f16) with f32 accumulation, register-prefetch
// pipelined LDS staging, 64x64 wave tiles (16 WMMA / wave / K-step).
// Per-expert: GEMM1 (+GELU, +prob scale) -> f16 hid in d_ws -> GEMM2 (+= out).
// ---------------------------------------------------------------------------

typedef __attribute__((ext_vector_type(16))) _Float16 v16h;
typedef __attribute__((ext_vector_type(8)))  _Float16 half8;
typedef __attribute__((ext_vector_type(4)))  _Float16 half4;
typedef __attribute__((ext_vector_type(4)))  float    vf4;
typedef __attribute__((ext_vector_type(8)))  float    v8f;

#define MOE_B   4096
#define MOE_E   8
#define MOE_IN  1024
#define MOE_H   2048
#define MOE_D   1024

#define BM 128      // block tile M
#define BN 128      // block tile N
#define BK 32       // K step (one WMMA K)
#define NTHREADS 128  // 4 waves, 2x2 grid of 64x64 wave tiles
#define LDSP 40     // LDS pitch in halfs (80 B, multiple of 16 B)

// -------------------------------- helpers ----------------------------------

__device__ __forceinline__ float moe_softmax_prob(const float* __restrict__ gate, int e) {
    float mx = gate[0];
#pragma unroll
    for (int j = 1; j < MOE_E; ++j) mx = fmaxf(mx, gate[j]);
    float s = 0.0f;
#pragma unroll
    for (int j = 0; j < MOE_E; ++j) s += expf(gate[j] - mx);
    return expf(gate[e] - mx) / s;
}

__device__ __forceinline__ v16h load_frag_a(const _Float16* __restrict__ base, int hi) {
    // a[p] = A[m][8*hi + p] (p<8), A[m][16 + 8*hi + (p-8)] (p>=8); base = &As[row*LDSP]
    half8 lo = *(const half8*)(base + 8 * hi);
    half8 hh = *(const half8*)(base + 16 + 8 * hi);
    return __builtin_shufflevector(lo, hh, 0, 1, 2, 3, 4, 5, 6, 7,
                                           8, 9, 10, 11, 12, 13, 14, 15);
}

__device__ __forceinline__ v16h load_frag_b(const _Float16* __restrict__ base, int hi) {
    // b[p] = B[8*hi + p][n] (p<8), B[16 + 8*hi + (p-8)][n] (p>=8); Bs stored N-major: base = &Bs[n*LDSP]
    half8 lo = *(const half8*)(base + 8 * hi);
    half8 hh = *(const half8*)(base + 16 + 8 * hi);
    return __builtin_shufflevector(lo, hh, 0, 1, 2, 3, 4, 5, 6, 7,
                                           8, 9, 10, 11, 12, 13, 14, 15);
}

// --------------------------- init: out = bias mix --------------------------

__global__ __launch_bounds__(256) void moe_init_out(const float* __restrict__ gate,
                                                    const float* __restrict__ b2,
                                                    float* __restrict__ out) {
    int i = blockIdx.x * blockDim.x + threadIdx.x;
    if (i >= MOE_B * MOE_D) return;
    int d = i & (MOE_D - 1);
    float mx = gate[0];
#pragma unroll
    for (int j = 1; j < MOE_E; ++j) mx = fmaxf(mx, gate[j]);
    float s = 0.0f;
    float p[MOE_E];
#pragma unroll
    for (int j = 0; j < MOE_E; ++j) { p[j] = expf(gate[j] - mx); s += p[j]; }
    float inv = 1.0f / s;
    float acc = 0.0f;
#pragma unroll
    for (int j = 0; j < MOE_E; ++j) acc += (p[j] * inv) * b2[j * MOE_D + d];
    out[i] = acc;
}

// ----------------- GEMM1: hid = probs[e]*GELU(h @ W1[e] + b1[e]) -----------
// A = h [B x IN] fp32 -> f16 while staging; B = W1[e] [IN x H] fp32 -> f16,
// transposed to N-major in LDS via packed half4 (ds_store_b64) writes.

__global__ __launch_bounds__(NTHREADS) void moe_gemm1(
        const float* __restrict__ hsrc, const float* __restrict__ W1e,
        const float* __restrict__ b1e, const float* __restrict__ gate,
        int expert, _Float16* __restrict__ hid) {
    __shared__ __align__(16) _Float16 As[BM * LDSP];
    __shared__ __align__(16) _Float16 Bs[BN * LDSP];

    const int tid  = threadIdx.x;
    const int lane = tid & 31;
    const int wave = tid >> 5;     // 0..3
    const int wm   = wave & 1;     // M offset wm*64
    const int wn   = wave >> 1;    // N offset wn*64
    const int m    = lane & 15;
    const int hi   = lane >> 4;
    const int bM0  = blockIdx.y * BM;
    const int n0   = blockIdx.x * BN;

    v8f acc[4][4];
#pragma unroll
    for (int i = 0; i < 4; ++i)
#pragma unroll
        for (int j = 0; j < 4; ++j) acc[i][j] = (v8f)0.0f;

    vf4 ra[8];        // A tile: 128x32 fp32 = 1024 float4, 8/thread
    vf4 rb[2][4];     // B tile: 32x128 fp32, each work item = 4(k) x 4(n) sub-tile

    auto loadA = [&](int k0) {
#pragma unroll
        for (int i = 0; i < 8; ++i) {
            int idx = tid + NTHREADS * i;     // float4 units, 8 per row
            int r   = idx >> 3;
            int c4  = idx & 7;
            ra[i] = *(const vf4*)(hsrc + (size_t)(bM0 + r) * MOE_IN + k0 + c4 * 4);
        }
    };
    auto loadB = [&](int k0) {
#pragma unroll
        for (int j = 0; j < 2; ++j) {
            int w  = tid + NTHREADS * j;      // 256 work items: 8 kquads x 32 n4-cols
            int kq = w >> 5;
            int n4 = w & 31;
#pragma unroll
            for (int jj = 0; jj < 4; ++jj)
                rb[j][jj] = *(const vf4*)(W1e + (size_t)(k0 + kq * 4 + jj) * MOE_H + n0 + n4 * 4);
        }
    };
    auto storeTiles = [&]() {
#pragma unroll
        for (int i = 0; i < 8; ++i) {
            int idx = tid + NTHREADS * i;
            int r   = idx >> 3;
            int c4  = idx & 7;
            half4 hv = { (_Float16)ra[i][0], (_Float16)ra[i][1],
                         (_Float16)ra[i][2], (_Float16)ra[i][3] };
            *(half4*)&As[r * LDSP + c4 * 4] = hv;
        }
#pragma unroll
        for (int j = 0; j < 2; ++j) {
            int w  = tid + NTHREADS * j;
            int kq = w >> 5;
            int n4 = w & 31;
#pragma unroll
            for (int i = 0; i < 4; ++i) {   // n within quad; 4 k-values packed per store
                half4 hv = { (_Float16)rb[j][0][i], (_Float16)rb[j][1][i],
                             (_Float16)rb[j][2][i], (_Float16)rb[j][3][i] };
                *(half4*)&Bs[(n4 * 4 + i) * LDSP + kq * 4] = hv;
            }
        }
    };

    loadA(0); loadB(0);
    for (int k0 = 0; k0 < MOE_IN; k0 += BK) {
        storeTiles();
        __syncthreads();
        if (k0 + BK < MOE_IN) { loadA(k0 + BK); loadB(k0 + BK); }  // overlap w/ compute

        v16h a[4], b[4];
#pragma unroll
        for (int mt = 0; mt < 4; ++mt)
            a[mt] = load_frag_a(&As[(wm * 64 + mt * 16 + m) * LDSP], hi);
#pragma unroll
        for (int nt = 0; nt < 4; ++nt)
            b[nt] = load_frag_b(&Bs[(wn * 64 + nt * 16 + m) * LDSP], hi);
#pragma unroll
        for (int mt = 0; mt < 4; ++mt)
#pragma unroll
            for (int nt = 0; nt < 4; ++nt)
                acc[mt][nt] = __builtin_amdgcn_wmma_f32_16x16x32_f16(
                    false, a[mt], false, b[nt], (short)0, acc[mt][nt], false, false);
        __syncthreads();
    }

    const float pe = moe_softmax_prob(gate, expert);

#pragma unroll
    for (int mt = 0; mt < 4; ++mt) {
#pragma unroll
        for (int nt = 0; nt < 4; ++nt) {
            const int row0 = bM0 + wm * 64 + mt * 16 + hi * 8;
            const int col  = n0 + wn * 64 + nt * 16 + m;
            const float bias = b1e[col];
#pragma unroll
            for (int r = 0; r < 8; ++r) {
                float v = acc[mt][nt][r] + bias;
                float g = 0.5f * v * (1.0f + erff(v * 0.70710678118f));
                hid[(size_t)(row0 + r) * MOE_H + col] = (_Float16)(pe * g);
            }
        }
    }
}

// ----------------- GEMM2: out += hid(f16, pre-scaled) @ W2[e] --------------

__global__ __launch_bounds__(NTHREADS) void moe_gemm2(
        const _Float16* __restrict__ hid, const float* __restrict__ W2e,
        float* __restrict__ out) {
    __shared__ __align__(16) _Float16 As[BM * LDSP];
    __shared__ __align__(16) _Float16 Bs[BN * LDSP];

    const int tid  = threadIdx.x;
    const int lane = tid & 31;
    const int wave = tid >> 5;
    const int wm   = wave & 1;
    const int wn   = wave >> 1;
    const int m    = lane & 15;
    const int hi   = lane >> 4;
    const int bM0  = blockIdx.y * BM;
    const int n0   = blockIdx.x * BN;

    v8f acc[4][4];
#pragma unroll
    for (int i = 0; i < 4; ++i)
#pragma unroll
        for (int j = 0; j < 4; ++j) acc[i][j] = (v8f)0.0f;

    half8 ra[4];      // A tile: 128x32 f16 = 512 half8, 4/thread
    vf4 rb[2][4];     // B tile: 32x128 fp32, 4(k) x 4(n) sub-tiles

    auto loadA = [&](int k0) {
#pragma unroll
        for (int i = 0; i < 4; ++i) {
            int idx = tid + NTHREADS * i;     // half8 units, 4 per row
            int r   = idx >> 2;
            int c8  = idx & 3;
            ra[i] = *(const half8*)(hid + (size_t)(bM0 + r) * MOE_H + k0 + c8 * 8);
        }
    };
    auto loadB = [&](int k0) {
#pragma unroll
        for (int j = 0; j < 2; ++j) {
            int w  = tid + NTHREADS * j;
            int kq = w >> 5;
            int n4 = w & 31;
#pragma unroll
            for (int jj = 0; jj < 4; ++jj)
                rb[j][jj] = *(const vf4*)(W2e + (size_t)(k0 + kq * 4 + jj) * MOE_D + n0 + n4 * 4);
        }
    };
    auto storeTiles = [&]() {
#pragma unroll
        for (int i = 0; i < 4; ++i) {
            int idx = tid + NTHREADS * i;
            int r   = idx >> 2;
            int c8  = idx & 3;
            *(half8*)&As[r * LDSP + c8 * 8] = ra[i];
        }
#pragma unroll
        for (int j = 0; j < 2; ++j) {
            int w  = tid + NTHREADS * j;
            int kq = w >> 5;
            int n4 = w & 31;
#pragma unroll
            for (int i = 0; i < 4; ++i) {
                half4 hv = { (_Float16)rb[j][0][i], (_Float16)rb[j][1][i],
                             (_Float16)rb[j][2][i], (_Float16)rb[j][3][i] };
                *(half4*)&Bs[(n4 * 4 + i) * LDSP + kq * 4] = hv;
            }
        }
    };

    loadA(0); loadB(0);
    for (int k0 = 0; k0 < MOE_H; k0 += BK) {
        storeTiles();
        __syncthreads();
        if (k0 + BK < MOE_H) { loadA(k0 + BK); loadB(k0 + BK); }

        v16h a[4], b[4];
#pragma unroll
        for (int mt = 0; mt < 4; ++mt)
            a[mt] = load_frag_a(&As[(wm * 64 + mt * 16 + m) * LDSP], hi);
#pragma unroll
        for (int nt = 0; nt < 4; ++nt)
            b[nt] = load_frag_b(&Bs[(wn * 64 + nt * 16 + m) * LDSP], hi);
#pragma unroll
        for (int mt = 0; mt < 4; ++mt)
#pragma unroll
            for (int nt = 0; nt < 4; ++nt)
                acc[mt][nt] = __builtin_amdgcn_wmma_f32_16x16x32_f16(
                    false, a[mt], false, b[nt], (short)0, acc[mt][nt], false, false);
        __syncthreads();
    }

    // epilogue: out += acc (one owner per element per launch; launches serialize)
#pragma unroll
    for (int mt = 0; mt < 4; ++mt) {
#pragma unroll
        for (int nt = 0; nt < 4; ++nt) {
            const int row0 = bM0 + wm * 64 + mt * 16 + hi * 8;
            const int col  = n0 + wn * 64 + nt * 16 + m;
#pragma unroll
            for (int r = 0; r < 8; ++r) {
                float* p = out + (size_t)(row0 + r) * MOE_D + col;
                *p = *p + acc[mt][nt][r];
            }
        }
    }
}

// --------------------------------- launch ----------------------------------

extern "C" void kernel_launch(void* const* d_in, const int* in_sizes, int n_in,
                              void* d_out, int out_size, void* d_ws, size_t ws_size,
                              hipStream_t stream) {
    const float* h    = (const float*)d_in[0];   // [B, IN]
    const float* gate = (const float*)d_in[1];   // [E]
    const float* W1   = (const float*)d_in[2];   // [E, IN, H]
    const float* b1   = (const float*)d_in[3];   // [E, H]
    const float* W2   = (const float*)d_in[4];   // [E, H, D]
    const float* b2   = (const float*)d_in[5];   // [E, D]
    float* out = (float*)d_out;                  // [B, D] fp32
    _Float16* hid = (_Float16*)d_ws;             // [B, H] f16 scratch (16 MB)

    (void)in_sizes; (void)n_in; (void)out_size; (void)ws_size;

    moe_init_out<<<(MOE_B * MOE_D + 255) / 256, 256, 0, stream>>>(gate, b2, out);

    for (int e = 0; e < MOE_E; ++e) {
        moe_gemm1<<<dim3(MOE_H / BN, MOE_B / BM), NTHREADS, 0, stream>>>(
            h, W1 + (size_t)e * MOE_IN * MOE_H, b1 + (size_t)e * MOE_H, gate, e, hid);
        moe_gemm2<<<dim3(MOE_D / BN, MOE_B / BM), NTHREADS, 0, stream>>>(
            hid, W2 + (size_t)e * MOE_H * MOE_D, out);
    }
}
